// GatedMLP_10093173146228
// MI455X (gfx1250) — compile-verified, matrix operand order checked by reading, not run on hardware
//
#include <hip/hip_runtime.h>
#include <hip/hip_bf16.h>

#define HIDDEN 4096
#define INTER  14336
#define TOKENS 4096

typedef __attribute__((ext_vector_type(16))) __bf16 v16bf;
typedef __attribute__((ext_vector_type(8)))  float  v8f;
typedef unsigned int __attribute__((ext_vector_type(4))) u32x4;
typedef int          __attribute__((ext_vector_type(4))) i32x4;
typedef int          __attribute__((ext_vector_type(8))) i32x8;

#if __has_builtin(__builtin_amdgcn_tensor_load_to_lds)
#define HAS_TDM 1
#else
#define HAS_TDM 0
#endif

// fp32 -> bf16 round-to-nearest-even
__device__ __forceinline__ unsigned short f32_bf16(float f) {
    unsigned int u = __float_as_uint(f);
    u += 0x7FFFu + ((u >> 16) & 1u);
    return (unsigned short)(u >> 16);
}
__device__ __forceinline__ unsigned int pack2_bf16(float a, float b) {
    return (unsigned int)f32_bf16(a) | ((unsigned int)f32_bf16(b) << 16);
}

__device__ __forceinline__ v8f wmma_bf16(v16bf a, v16bf b, v8f c) {
    return __builtin_amdgcn_wmma_f32_16x16x32_bf16(false, a, false, b,
                                                   (short)0, c, false, false);
}

#if HAS_TDM
// -------------------------------------------------------------------------
// TDM: 2D bf16 tile load (tile1 rows x tile0 elems) from a row-major
// [dim1, dim0] tensor (row stride = stride0 elems) into LDS at lds_addr.
// D# packing per CDNA5 ISA.
// -------------------------------------------------------------------------
__device__ __forceinline__ void tdm_load_2d_bf16(
    unsigned lds_addr, unsigned long long ga,
    unsigned dim0, unsigned dim1,
    unsigned tile0, unsigned tile1, unsigned long long stride0)
{
    u32x4 g0;
    g0.x = 1u;                                               // count=1 (valid)
    g0.y = lds_addr;                                         // lds_addr[31:0]
    g0.z = (unsigned)(ga & 0xFFFFFFFFu);                     // global_addr lo
    g0.w = (unsigned)((ga >> 32) & 0x1FFFFFFu) | (2u << 30); // addr hi | type=2
    i32x8 g1;
    g1[0] = (int)(1u << 16);                                 // data_size=1 -> 2B
    g1[1] = (int)((dim0 & 0xFFFFu) << 16);                   // tensor_dim0 lo
    g1[2] = (int)(((dim0 >> 16) & 0xFFFFu) | ((dim1 & 0xFFFFu) << 16));
    g1[3] = (int)(((dim1 >> 16) & 0xFFFFu) | ((tile0 & 0xFFFFu) << 16));
    g1[4] = (int)(tile1 & 0xFFFFu);                          // tile_dim1; tile_dim2=0
    g1[5] = (int)(stride0 & 0xFFFFFFFFull);                  // dim0_stride lo32
    g1[6] = (int)((stride0 >> 32) & 0xFFFFull);              // dim0_stride hi16
    g1[7] = 0;
    const i32x4 z4 = {0, 0, 0, 0};
#if __has_include(<hip/amd_detail/amd_gfx1250_TDM.h>)
    const i32x8 z8 = {0, 0, 0, 0, 0, 0, 0, 0};
    __builtin_amdgcn_tensor_load_to_lds(g0, g1, z4, z4, z8, 0);
#else
    __builtin_amdgcn_tensor_load_to_lds(g0, g1, z4, z4, 0);
#endif
}
#endif // HAS_TDM

__device__ __forceinline__ unsigned lds_off(const void* p) {
    return (unsigned)(uintptr_t)p;  // LDS aperture: addr[31:0] = LDS byte offset
}

// ===========================================================================
// Pre-pass: fp32 -> bf16 elementwise (memory-bound; off the GEMM hot loop)
// ===========================================================================
__global__ __launch_bounds__(256)
void f32_to_bf16_kernel(const float* __restrict__ in,
                        unsigned short* __restrict__ out, long n)
{
    long i = ((long)blockIdx.x * 256 + threadIdx.x) * 8;
    const long stride = (long)gridDim.x * 256 * 8;
    for (; i < n; i += stride) {
        const float4 a = *reinterpret_cast<const float4*>(in + i);
        const float4 b = *reinterpret_cast<const float4*>(in + i + 4);
        uint4 o;
        o.x = pack2_bf16(a.x, a.y);
        o.y = pack2_bf16(a.z, a.w);
        o.z = pack2_bf16(b.x, b.y);
        o.w = pack2_bf16(b.z, b.w);
        *reinterpret_cast<uint4*>(out + i) = o;
    }
}

// ===========================================================================
// Path A kernel 1: gate/up GEMM, pure bf16, TDM-staged K=64 slices,
// double-buffered. SiLU(gate)*up epilogue -> bf16 h2.
// Block tile 128 x 64 (+64 up). 8 waves = 2(m) x 4(n). 16 WMMA / stage / wave.
// LDS: A 2x16KB + Bg 2x8KB + Bu 2x8KB = 64 KB.
// ===========================================================================
__global__ __launch_bounds__(256)
void gemm_gateup_bf16(const unsigned short* __restrict__ xb,  // bf16 [T,H]
                      const unsigned short* __restrict__ wb,  // bf16 [2I,H]
                      unsigned short* __restrict__ h2)        // bf16 [T,I]
{
    __shared__ alignas(32) unsigned short sA [2][128 * 64];
    __shared__ alignas(32) unsigned short sBg[2][ 64 * 64];
    __shared__ alignas(32) unsigned short sBu[2][ 64 * 64];

    const int tid  = threadIdx.x;
    const int lane = tid & 31;
    const int wave = tid >> 5;
    const int wm   = wave >> 2;
    const int wn   = wave &  3;
    const int t0 = blockIdx.x * 128;
    const int n0 = blockIdx.y * 64;

    v8f accG[4], accU[4];
    #pragma unroll
    for (int i = 0; i < 4; ++i)
        #pragma unroll
        for (int j = 0; j < 8; ++j) { accG[i][j] = 0.0f; accU[i][j] = 0.0f; }

    const int frow  = lane & 15;
    const int fkoff = (lane >> 4) * 16;

#if HAS_TDM
    // uniform global tile addresses, stepped by 128B (64 bf16) per stage
    unsigned long long gaA = (unsigned long long)(uintptr_t)&xb[(size_t)t0 * HIDDEN];
    unsigned long long gaG = (unsigned long long)(uintptr_t)&wb[(size_t)n0 * HIDDEN];
    unsigned long long gaU = (unsigned long long)(uintptr_t)&wb[(size_t)(INTER + n0) * HIDDEN];

    if (tid < 32) {
        tdm_load_2d_bf16(lds_off(&sA [0][0]), gaA, HIDDEN, TOKENS,   64, 128, HIDDEN);
        tdm_load_2d_bf16(lds_off(&sBg[0][0]), gaG, HIDDEN, 2 * INTER, 64, 64, HIDDEN);
        tdm_load_2d_bf16(lds_off(&sBu[0][0]), gaU, HIDDEN, 2 * INTER, 64, 64, HIDDEN);
    }
    const int NSTAGE = HIDDEN / 64;
    for (int it = 0; it < NSTAGE; ++it) {
        const int cur = it & 1;
        const bool more = (it + 1) < NSTAGE;
        gaA += 128; gaG += 128; gaU += 128;   // next K slice (64 elems * 2B)
        if (tid < 32) {
            if (more) {
                tdm_load_2d_bf16(lds_off(&sA [cur ^ 1][0]), gaA,
                                 HIDDEN, TOKENS,   64, 128, HIDDEN);
                tdm_load_2d_bf16(lds_off(&sBg[cur ^ 1][0]), gaG,
                                 HIDDEN, 2 * INTER, 64, 64, HIDDEN);
                tdm_load_2d_bf16(lds_off(&sBu[cur ^ 1][0]), gaU,
                                 HIDDEN, 2 * INTER, 64, 64, HIDDEN);
                __builtin_amdgcn_s_wait_tensorcnt(3);   // stage `cur` complete
            } else {
                __builtin_amdgcn_s_wait_tensorcnt(0);
            }
        }
        __syncthreads();

        #pragma unroll
        for (int kc = 0; kc < 64; kc += 32) {
            const v16bf bg = *reinterpret_cast<const v16bf*>(
                &sBg[cur][(wn * 16 + frow) * 64 + kc + fkoff]);
            const v16bf bu = *reinterpret_cast<const v16bf*>(
                &sBu[cur][(wn * 16 + frow) * 64 + kc + fkoff]);
            #pragma unroll
            for (int tm = 0; tm < 4; ++tm) {
                const v16bf a = *reinterpret_cast<const v16bf*>(
                    &sA[cur][(wm * 64 + tm * 16 + frow) * 64 + kc + fkoff]);
                accG[tm] = wmma_bf16(a, bg, accG[tm]);
                accU[tm] = wmma_bf16(a, bu, accU[tm]);
            }
        }
        __syncthreads();
    }
#else
    // fallback: cooperative bf16 tile copies (no converts), single buffer
    const int crow = tid >> 1;              // 0..127
    const int ccol = (tid & 1) * 32;        // bf16 elems
    for (int kk = 0; kk < HIDDEN; kk += 64) {
        #pragma unroll
        for (int q = 0; q < 4; ++q)
            *reinterpret_cast<uint4*>(&sA[0][crow * 64 + ccol + q * 8]) =
                *reinterpret_cast<const uint4*>(
                    &xb[(size_t)(t0 + crow) * HIDDEN + kk + ccol + q * 8]);
        if (crow < 64) {
            #pragma unroll
            for (int q = 0; q < 4; ++q)
                *reinterpret_cast<uint4*>(&sBg[0][crow * 64 + ccol + q * 8]) =
                    *reinterpret_cast<const uint4*>(
                        &wb[(size_t)(n0 + crow) * HIDDEN + kk + ccol + q * 8]);
        } else {
            const int r = crow - 64;
            #pragma unroll
            for (int q = 0; q < 4; ++q)
                *reinterpret_cast<uint4*>(&sBu[0][r * 64 + ccol + q * 8]) =
                    *reinterpret_cast<const uint4*>(
                        &wb[(size_t)(INTER + n0 + r) * HIDDEN + kk + ccol + q * 8]);
        }
        __syncthreads();
        #pragma unroll
        for (int kc = 0; kc < 64; kc += 32) {
            const v16bf bg = *reinterpret_cast<const v16bf*>(
                &sBg[0][(wn * 16 + frow) * 64 + kc + fkoff]);
            const v16bf bu = *reinterpret_cast<const v16bf*>(
                &sBu[0][(wn * 16 + frow) * 64 + kc + fkoff]);
            #pragma unroll
            for (int tm = 0; tm < 4; ++tm) {
                const v16bf a = *reinterpret_cast<const v16bf*>(
                    &sA[0][(wm * 64 + tm * 16 + frow) * 64 + kc + fkoff]);
                accG[tm] = wmma_bf16(a, bg, accG[tm]);
                accU[tm] = wmma_bf16(a, bu, accU[tm]);
            }
        }
        __syncthreads();
    }
#endif

    const int cm = (lane >> 4) * 8;
    const int cn = lane & 15;
    #pragma unroll
    for (int tm = 0; tm < 4; ++tm) {
        #pragma unroll
        for (int j = 0; j < 8; ++j) {
            const float g = accG[tm][j];
            const float u = accU[tm][j];
            const float s = g / (1.0f + __expf(-g));
            const int m = t0 + wm * 64 + tm * 16 + cm + j;
            const int n = n0 + wn * 16 + cn;
            h2[(size_t)m * INTER + n] = f32_bf16(s * u);
        }
    }
}

// ===========================================================================
// Path A kernel 2: out = h2 @ w_down^T, pure bf16, TDM-staged K=64 slices,
// double-buffered. Block tile 128 x 128. 8 waves = 2(m) x 4(n).
// 16 WMMA / stage / wave. LDS: A 2x16KB + B 2x16KB = 64 KB.
// ===========================================================================
__global__ __launch_bounds__(256)
void gemm_down_bf16(const unsigned short* __restrict__ h2,   // bf16 [T,I]
                    const unsigned short* __restrict__ wdb,  // bf16 [H,I]
                    float* __restrict__ out)                 // f32  [T,H]
{
    __shared__ alignas(32) unsigned short sA[2][128 * 64];
    __shared__ alignas(32) unsigned short sB[2][128 * 64];

    const int tid  = threadIdx.x;
    const int lane = tid & 31;
    const int wave = tid >> 5;
    const int wm   = wave >> 2;
    const int wn   = wave &  3;
    const int t0 = blockIdx.x * 128;
    const int h0 = blockIdx.y * 128;

    v8f acc[4][2];
    #pragma unroll
    for (int i = 0; i < 4; ++i)
        #pragma unroll
        for (int t = 0; t < 2; ++t)
            #pragma unroll
            for (int j = 0; j < 8; ++j) acc[i][t][j] = 0.0f;

    const int frow  = lane & 15;
    const int fkoff = (lane >> 4) * 16;

#if HAS_TDM
    unsigned long long gaA = (unsigned long long)(uintptr_t)&h2[(size_t)t0 * INTER];
    unsigned long long gaB = (unsigned long long)(uintptr_t)&wdb[(size_t)h0 * INTER];

    if (tid < 32) {
        tdm_load_2d_bf16(lds_off(&sA[0][0]), gaA, INTER, TOKENS, 64, 128, INTER);
        tdm_load_2d_bf16(lds_off(&sB[0][0]), gaB, INTER, HIDDEN, 64, 128, INTER);
    }
    const int NSTAGE = INTER / 64;
    for (int it = 0; it < NSTAGE; ++it) {
        const int cur = it & 1;
        const bool more = (it + 1) < NSTAGE;
        gaA += 128; gaB += 128;
        if (tid < 32) {
            if (more) {
                tdm_load_2d_bf16(lds_off(&sA[cur ^ 1][0]), gaA,
                                 INTER, TOKENS, 64, 128, INTER);
                tdm_load_2d_bf16(lds_off(&sB[cur ^ 1][0]), gaB,
                                 INTER, HIDDEN, 64, 128, INTER);
                __builtin_amdgcn_s_wait_tensorcnt(2);
            } else {
                __builtin_amdgcn_s_wait_tensorcnt(0);
            }
        }
        __syncthreads();

        #pragma unroll
        for (int kc = 0; kc < 64; kc += 32) {
            v16bf bfrag[2];
            #pragma unroll
            for (int tn = 0; tn < 2; ++tn)
                bfrag[tn] = *reinterpret_cast<const v16bf*>(
                    &sB[cur][(wn * 32 + tn * 16 + frow) * 64 + kc + fkoff]);
            #pragma unroll
            for (int tm = 0; tm < 4; ++tm) {
                const v16bf a = *reinterpret_cast<const v16bf*>(
                    &sA[cur][(wm * 64 + tm * 16 + frow) * 64 + kc + fkoff]);
                #pragma unroll
                for (int tn = 0; tn < 2; ++tn)
                    acc[tm][tn] = wmma_bf16(a, bfrag[tn], acc[tm][tn]);
            }
        }
        __syncthreads();
    }
#else
    const int crow = tid >> 1;
    const int ccol = (tid & 1) * 32;
    for (int kk = 0; kk < INTER; kk += 64) {
        #pragma unroll
        for (int q = 0; q < 4; ++q) {
            *reinterpret_cast<uint4*>(&sA[0][crow * 64 + ccol + q * 8]) =
                *reinterpret_cast<const uint4*>(
                    &h2[(size_t)(t0 + crow) * INTER + kk + ccol + q * 8]);
            *reinterpret_cast<uint4*>(&sB[0][crow * 64 + ccol + q * 8]) =
                *reinterpret_cast<const uint4*>(
                    &wdb[(size_t)(h0 + crow) * INTER + kk + ccol + q * 8]);
        }
        __syncthreads();
        #pragma unroll
        for (int kc = 0; kc < 64; kc += 32) {
            v16bf bfrag[2];
            #pragma unroll
            for (int tn = 0; tn < 2; ++tn)
                bfrag[tn] = *reinterpret_cast<const v16bf*>(
                    &sB[0][(wn * 32 + tn * 16 + frow) * 64 + kc + fkoff]);
            #pragma unroll
            for (int tm = 0; tm < 4; ++tm) {
                const v16bf a = *reinterpret_cast<const v16bf*>(
                    &sA[0][(wm * 64 + tm * 16 + frow) * 64 + kc + fkoff]);
                #pragma unroll
                for (int tn = 0; tn < 2; ++tn)
                    acc[tm][tn] = wmma_bf16(a, bfrag[tn], acc[tm][tn]);
            }
        }
        __syncthreads();
    }
#endif

    const int cm = (lane >> 4) * 8;
    const int cn = lane & 15;
    #pragma unroll
    for (int tm = 0; tm < 4; ++tm)
        #pragma unroll
        for (int tn = 0; tn < 2; ++tn)
            #pragma unroll
            for (int j = 0; j < 8; ++j) {
                const int m = t0 + wm * 64 + tm * 16 + cm + j;
                const int n = h0 + wn * 32 + tn * 16 + cn;
                out[(size_t)m * HIDDEN + n] = acc[tm][tn][j];
            }
}

// ===========================================================================
// Path B (small workspace fallback): fused-convert GEMMs.
// ===========================================================================
__global__ __launch_bounds__(256)
void mlp_gateup_silu_fused(const float* __restrict__ x,
                           const float* __restrict__ w,
                           unsigned short* __restrict__ h2)
{
    __shared__ alignas(32) unsigned short sA [128 * 32];
    __shared__ alignas(32) unsigned short sBg[ 64 * 32];
    __shared__ alignas(32) unsigned short sBu[ 64 * 32];

    const int tid  = threadIdx.x;
    const int lane = tid & 31;
    const int wave = tid >> 5;
    const int wm   = wave >> 2;
    const int wn   = wave &  3;
    const int t0 = blockIdx.x * 128;
    const int n0 = blockIdx.y * 64;

    v8f accG[4], accU[4];
    #pragma unroll
    for (int i = 0; i < 4; ++i)
        #pragma unroll
        for (int j = 0; j < 8; ++j) { accG[i][j] = 0.0f; accU[i][j] = 0.0f; }

    const int frow  = lane & 15;
    const int fkoff = (lane >> 4) * 16;
    const int lrow = tid >> 3;
    const int lcol = (tid & 7) * 4;

    for (int kk = 0; kk < HIDDEN; kk += 32) {
        #pragma unroll
        for (int p = 0; p < 4; ++p) {
            const int r = lrow + p * 32;
            const float4 v = *reinterpret_cast<const float4*>(
                &x[(size_t)(t0 + r) * HIDDEN + kk + lcol]);
            *reinterpret_cast<uint2*>(&sA[r * 32 + lcol]) =
                make_uint2(pack2_bf16(v.x, v.y), pack2_bf16(v.z, v.w));
        }
        #pragma unroll
        for (int p = 0; p < 2; ++p) {
            const int r = lrow + p * 32;
            {
                const float4 v = *reinterpret_cast<const float4*>(
                    &w[(size_t)(n0 + r) * HIDDEN + kk + lcol]);
                *reinterpret_cast<uint2*>(&sBg[r * 32 + lcol]) =
                    make_uint2(pack2_bf16(v.x, v.y), pack2_bf16(v.z, v.w));
            }
            {
                const float4 v = *reinterpret_cast<const float4*>(
                    &w[(size_t)(INTER + n0 + r) * HIDDEN + kk + lcol]);
                *reinterpret_cast<uint2*>(&sBu[r * 32 + lcol]) =
                    make_uint2(pack2_bf16(v.x, v.y), pack2_bf16(v.z, v.w));
            }
        }
        __syncthreads();
        const v16bf bg = *reinterpret_cast<const v16bf*>(
            &sBg[(wn * 16 + frow) * 32 + fkoff]);
        const v16bf bu = *reinterpret_cast<const v16bf*>(
            &sBu[(wn * 16 + frow) * 32 + fkoff]);
        #pragma unroll
        for (int tm = 0; tm < 4; ++tm) {
            const v16bf a = *reinterpret_cast<const v16bf*>(
                &sA[(wm * 64 + tm * 16 + frow) * 32 + fkoff]);
            accG[tm] = wmma_bf16(a, bg, accG[tm]);
            accU[tm] = wmma_bf16(a, bu, accU[tm]);
        }
        __syncthreads();
    }

    const int cm = (lane >> 4) * 8;
    const int cn = lane & 15;
    #pragma unroll
    for (int tm = 0; tm < 4; ++tm)
        #pragma unroll
        for (int j = 0; j < 8; ++j) {
            const float g = accG[tm][j];
            const float u = accU[tm][j];
            const float s = g / (1.0f + __expf(-g));
            const int m = t0 + wm * 64 + tm * 16 + cm + j;
            const int n = n0 + wn * 16 + cn;
            h2[(size_t)m * INTER + n] = f32_bf16(s * u);
        }
}

__global__ __launch_bounds__(256)
void mlp_down_fused(const unsigned short* __restrict__ h2,
                    const float* __restrict__ wd,
                    float* __restrict__ out)
{
    __shared__ alignas(32) unsigned short sA[128 * 32];
    __shared__ alignas(32) unsigned short sB[128 * 32];

    const int tid  = threadIdx.x;
    const int lane = tid & 31;
    const int wave = tid >> 5;
    const int wm   = wave >> 2;
    const int wn   = wave &  3;
    const int t0 = blockIdx.x * 128;
    const int h0 = blockIdx.y * 128;

    v8f acc[4][2];
    #pragma unroll
    for (int i = 0; i < 4; ++i)
        #pragma unroll
        for (int t = 0; t < 2; ++t)
            #pragma unroll
            for (int j = 0; j < 8; ++j) acc[i][t][j] = 0.0f;

    const int frow  = lane & 15;
    const int fkoff = (lane >> 4) * 16;
    const int a_row = tid >> 1;
    const int a_col = (tid & 1) * 16;
    const int b_row = tid >> 3;
    const int b_col = (tid & 7) * 4;

    for (int kk = 0; kk < INTER; kk += 32) {
        *reinterpret_cast<uint4*>(&sA[a_row * 32 + a_col]) =
            *reinterpret_cast<const uint4*>(&h2[(size_t)(t0 + a_row) * INTER + kk + a_col]);
        *reinterpret_cast<uint4*>(&sA[a_row * 32 + a_col + 8]) =
            *reinterpret_cast<const uint4*>(&h2[(size_t)(t0 + a_row) * INTER + kk + a_col + 8]);
        #pragma unroll
        for (int p = 0; p < 4; ++p) {
            const int r = b_row + p * 32;
            const float4 v = *reinterpret_cast<const float4*>(
                &wd[(size_t)(h0 + r) * INTER + kk + b_col]);
            *reinterpret_cast<uint2*>(&sB[r * 32 + b_col]) =
                make_uint2(pack2_bf16(v.x, v.y), pack2_bf16(v.z, v.w));
        }
        __syncthreads();
        v16bf bfrag[2];
        #pragma unroll
        for (int tn = 0; tn < 2; ++tn)
            bfrag[tn] = *reinterpret_cast<const v16bf*>(
                &sB[(wn * 32 + tn * 16 + frow) * 32 + fkoff]);
        #pragma unroll
        for (int tm = 0; tm < 4; ++tm) {
            const v16bf a = *reinterpret_cast<const v16bf*>(
                &sA[(wm * 64 + tm * 16 + frow) * 32 + fkoff]);
            #pragma unroll
            for (int tn = 0; tn < 2; ++tn)
                acc[tm][tn] = wmma_bf16(a, bfrag[tn], acc[tm][tn]);
        }
        __syncthreads();
    }

    const int cm = (lane >> 4) * 8;
    const int cn = lane & 15;
    #pragma unroll
    for (int tm = 0; tm < 4; ++tm)
        #pragma unroll
        for (int tn = 0; tn < 2; ++tn)
            #pragma unroll
            for (int j = 0; j < 8; ++j) {
                const int m = t0 + wm * 64 + tm * 16 + cm + j;
                const int n = h0 + wn * 32 + tn * 16 + cn;
                out[(size_t)m * HIDDEN + n] = acc[tm][tn][j];
            }
}

extern "C" void kernel_launch(void* const* d_in, const int* in_sizes, int n_in,
                              void* d_out, int out_size, void* d_ws, size_t ws_size,
                              hipStream_t stream) {
    const float* x         = (const float*)d_in[0];  // [TOKENS, HIDDEN]
    const float* w_gate_up = (const float*)d_in[1];  // [2*INTER, HIDDEN]
    const float* w_down    = (const float*)d_in[2];  // [HIDDEN, INTER]
    float* out = (float*)d_out;

    const size_t nx  = (size_t)TOKENS * HIDDEN;
    const size_t nw1 = (size_t)2 * INTER * HIDDEN;
    const size_t nw2 = (size_t)HIDDEN * INTER;
    const size_t nh2 = (size_t)TOKENS * INTER;
    const size_t needA = 2 * (nx + nw1 + nw2 + nh2);   // ~480 MB

    if (ws_size >= needA) {
        // Path A: one-time bf16 preconversion, then pure-bf16 GEMMs (TDM staged)
        unsigned short* xb  = (unsigned short*)d_ws;
        unsigned short* wb  = xb + nx;
        unsigned short* wdb = wb + nw1;
        unsigned short* h2  = wdb + nw2;
        f32_to_bf16_kernel<<<8192, 256, 0, stream>>>(x, xb, (long)nx);
        f32_to_bf16_kernel<<<8192, 256, 0, stream>>>(w_gate_up, wb, (long)nw1);
        f32_to_bf16_kernel<<<8192, 256, 0, stream>>>(w_down, wdb, (long)nw2);
        gemm_gateup_bf16<<<dim3(TOKENS / 128, INTER / 64), 256, 0, stream>>>(xb, wb, h2);
        gemm_down_bf16<<<dim3(TOKENS / 128, HIDDEN / 128), 256, 0, stream>>>(h2, wdb, out);
    } else {
        // Path B: fused-convert GEMMs, only h2 (112 MB) in workspace
        unsigned short* h2 = (unsigned short*)d_ws;
        mlp_gateup_silu_fused<<<dim3(TOKENS / 128, INTER / 64), 256, 0, stream>>>(x, w_gate_up, h2);
        mlp_down_fused<<<dim3(TOKENS / 128, HIDDEN / 128), 256, 0, stream>>>(h2, w_down, out);
    }
}
